// Loss_meta_25778393711118
// MI455X (gfx1250) — compile-verified, matrix operand order checked by reading, not run on hardware
//
#include <hip/hip_runtime.h>
#include <hip/hip_bf16.h>
#include <math.h>

#define N_SMP 8192
#define C_CLS 1000
#define A_DIM 2048
#define CP    1024            // padded class count (multiple of 64)
#define K2    (2 * A_DIM)     // 4096 : concatenated [term1 | term3] K
#define KP    (3 * K2)        // 12288: bf16x3 compensated K

typedef __bf16 bf16_t;
typedef __attribute__((ext_vector_type(16))) __bf16 v16bf;
typedef __attribute__((ext_vector_type(8)))  float  v8f;

// split f32 into bf16 hi + bf16 lo (x ~= hi + lo)
__device__ __forceinline__ void bsplit(float x, bf16_t& h, bf16_t& l) {
  h = (bf16_t)x;
  l = (bf16_t)(x - (float)h);
}

// ---------------------------------------------------------------------------
// Kernel 1: build packed bf16x3 operands + per-class term2.
//   A'[y] = [ cv[y,:] , -2*fc[y,:]*cv[y,:] ]   (length K2)
//   B'[j] = [ fc[j,:]^2 , fc[j,:] ]            (length K2)
//   Apack row = [hi(A') | hi(A') | lo(A')], Bpack row = [hi(B') | lo(B') | hi(B')]
//   => dot(Apack,Bpack) = sum hi*hi + hi*lo + lo*hi  ~= f32 dot(A',B')
// ---------------------------------------------------------------------------
__global__ void msaug_pack_kernel(const float* __restrict__ fc,
                                  const float* __restrict__ cv,
                                  bf16_t* __restrict__ Ap,
                                  bf16_t* __restrict__ Bp,
                                  float* __restrict__ t2) {
  const int c = blockIdx.x;
  const bool valid = c < C_CLS;
  const size_t row = (size_t)c * KP;
  float acc = 0.f;

  for (int a = threadIdx.x; a < A_DIM; a += blockDim.x) {
    float f = 0.f, v = 0.f;
    if (valid) {
      f = fc[(size_t)c * A_DIM + a];
      v = cv[(size_t)c * A_DIM + a];
    }
    acc += f * f * v;

    const float p0 = v;             // A' at k = a
    const float p1 = -2.f * f * v;  // A' at k = A_DIM + a
    const float q0 = f * f;         // B' at k = a
    const float q1 = f;             // B' at k = A_DIM + a

    bf16_t h, l;
    bsplit(p0, h, l);
    Ap[row + a] = h;               Ap[row + K2 + a] = h;               Ap[row + 2*K2 + a] = l;
    bsplit(p1, h, l);
    Ap[row + A_DIM + a] = h;       Ap[row + K2 + A_DIM + a] = h;       Ap[row + 2*K2 + A_DIM + a] = l;
    bsplit(q0, h, l);
    Bp[row + a] = h;               Bp[row + 2*K2 + a] = h;             Bp[row + K2 + a] = l;
    bsplit(q1, h, l);
    Bp[row + A_DIM + a] = h;       Bp[row + 2*K2 + A_DIM + a] = h;     Bp[row + K2 + A_DIM + a] = l;
  }

  __shared__ float red[256];
  red[threadIdx.x] = acc;
  __syncthreads();
  for (int s = 128; s > 0; s >>= 1) {
    if (threadIdx.x < s) red[threadIdx.x] += red[threadIdx.x + s];
    __syncthreads();
  }
  if (threadIdx.x == 0) t2[c] = red[0];
}

// ---------------------------------------------------------------------------
// Kernel 2: S[y,j] = 0.5*ratio*( dot(Apack[y], Bpack[j]) + t2[y] )
// 16x16 blocks of (4 waves); each wave computes a 32x32 output tile with
// 2x2 v_wmma_f32_16x16x32_bf16 fragments. Operands stream from L2.
// ---------------------------------------------------------------------------
__global__ void __launch_bounds__(128)
msaug_gemm_kernel(const bf16_t* __restrict__ Ap,
                  const bf16_t* __restrict__ Bp,
                  const float* __restrict__ t2,
                  const float* __restrict__ ratio,
                  float* __restrict__ S) {
  const int lane = threadIdx.x & 31;
  const int wave = threadIdx.x >> 5;
  const int wy = wave >> 1, wx = wave & 1;
  const int y0 = blockIdx.y * 64 + wy * 32;
  const int j0 = blockIdx.x * 64 + wx * 32;
  const int l  = lane & 15;
  const int hh = lane >> 4;          // which 16-element K chunk this lane holds

  v8f acc00 = {}, acc01 = {}, acc10 = {}, acc11 = {};

  const bf16_t* a0p = Ap + (size_t)(y0 + l)      * KP + hh * 16;
  const bf16_t* a1p = Ap + (size_t)(y0 + 16 + l) * KP + hh * 16;
  const bf16_t* b0p = Bp + (size_t)(j0 + l)      * KP + hh * 16;
  const bf16_t* b1p = Bp + (size_t)(j0 + 16 + l) * KP + hh * 16;

  #pragma unroll 2
  for (int k = 0; k < KP; k += 32) {
    v16bf a0 = *(const v16bf*)(a0p + k);   // 32B aligned
    v16bf a1 = *(const v16bf*)(a1p + k);
    v16bf b0 = *(const v16bf*)(b0p + k);
    v16bf b1 = *(const v16bf*)(b1p + k);
    acc00 = __builtin_amdgcn_wmma_f32_16x16x32_bf16(false, a0, false, b0, (short)0, acc00, false, false);
    acc01 = __builtin_amdgcn_wmma_f32_16x16x32_bf16(false, a0, false, b1, (short)0, acc01, false, false);
    acc10 = __builtin_amdgcn_wmma_f32_16x16x32_bf16(false, a1, false, b0, (short)0, acc10, false, false);
    acc11 = __builtin_amdgcn_wmma_f32_16x16x32_bf16(false, a1, false, b1, (short)0, acc11, false, false);
  }

  const float scale = 0.5f * ratio[0];
  // C/D layout: lane&15 = N column, VGPR r -> M = r + (lane>>4)*8
  #pragma unroll
  for (int r = 0; r < 8; ++r) {
    const int m = hh * 8 + r;
    {
      int y = y0 + m;
      S[(size_t)y * CP + (j0 + l)]      = scale * (acc00[r] + t2[y]);
      S[(size_t)y * CP + (j0 + 16 + l)] = scale * (acc01[r] + t2[y]);
    }
    {
      int y = y0 + 16 + m;
      S[(size_t)y * CP + (j0 + l)]      = scale * (acc10[r] + t2[y]);
      S[(size_t)y * CP + (j0 + 16 + l)] = scale * (acc11[r] + t2[y]);
    }
  }
}

// ---------------------------------------------------------------------------
// Kernel 3: per-sample weighted NLL over aug logits = y_s[i,:] + S[label_i,:]
// One block per sample; fixed-order block reductions (deterministic).
// ---------------------------------------------------------------------------
__global__ void msaug_loss_kernel(const float* __restrict__ y_s,
                                  const int* __restrict__ labels,
                                  const float* __restrict__ S,
                                  const float* __restrict__ weights,
                                  float* __restrict__ wn,
                                  float* __restrict__ wd) {
  const int i = blockIdx.x;
  const int lab = labels[i];
  const float* ys = y_s + (size_t)i * C_CLS;
  const float* sr = S + (size_t)lab * CP;

  __shared__ float red[256];
  __shared__ float s_lab;

  float vals[4];
  float lmax = -INFINITY;
  #pragma unroll
  for (int t = 0; t < 4; ++t) {
    const int j = threadIdx.x + t * 256;
    float v = -INFINITY;
    if (j < C_CLS) {
      v = ys[j] + sr[j];
      if (j == lab) s_lab = v;
    }
    vals[t] = v;
    lmax = fmaxf(lmax, v);
  }
  red[threadIdx.x] = lmax;
  __syncthreads();
  for (int s = 128; s > 0; s >>= 1) {
    if (threadIdx.x < s) red[threadIdx.x] = fmaxf(red[threadIdx.x], red[threadIdx.x + s]);
    __syncthreads();
  }
  const float m = red[0];
  __syncthreads();

  float lsum = 0.f;
  #pragma unroll
  for (int t = 0; t < 4; ++t)
    if (vals[t] > -INFINITY) lsum += expf(vals[t] - m);
  red[threadIdx.x] = lsum;
  __syncthreads();
  for (int s = 128; s > 0; s >>= 1) {
    if (threadIdx.x < s) red[threadIdx.x] += red[threadIdx.x + s];
    __syncthreads();
  }
  if (threadIdx.x == 0) {
    const float nll = logf(red[0]) + m - s_lab;
    const float w = weights[lab];
    wn[i] = w * nll;
    wd[i] = w;
  }
}

// ---------------------------------------------------------------------------
// Kernel 4: deterministic final reduction -> scalar loss
// ---------------------------------------------------------------------------
__global__ void msaug_final_kernel(const float* __restrict__ wn,
                                   const float* __restrict__ wd,
                                   float* __restrict__ out) {
  __shared__ float rn[256];
  __shared__ float rd[256];
  float sn = 0.f, sd = 0.f;
  for (int i = threadIdx.x; i < N_SMP; i += 256) {
    sn += wn[i];
    sd += wd[i];
  }
  rn[threadIdx.x] = sn;
  rd[threadIdx.x] = sd;
  __syncthreads();
  for (int s = 128; s > 0; s >>= 1) {
    if (threadIdx.x < s) {
      rn[threadIdx.x] += rn[threadIdx.x + s];
      rd[threadIdx.x] += rd[threadIdx.x + s];
    }
    __syncthreads();
  }
  if (threadIdx.x == 0) out[0] = rn[0] / rd[0];
}

// ---------------------------------------------------------------------------
extern "C" void kernel_launch(void* const* d_in, const int* in_sizes, int n_in,
                              void* d_out, int out_size, void* d_ws, size_t ws_size,
                              hipStream_t stream) {
  const float* fc      = (const float*)d_in[0];
  // d_in[1] features_source: unused by this loss path
  const float* y_s     = (const float*)d_in[2];
  const int*   labels  = (const int*)d_in[3];
  const float* ratio   = (const float*)d_in[4];
  const float* weights = (const float*)d_in[5];
  const float* cv      = (const float*)d_in[6];

  char* ws = (char*)d_ws;
  const size_t apBytes = (size_t)CP * KP * sizeof(bf16_t);   // 25,165,824
  bf16_t* Ap = (bf16_t*)(ws);
  bf16_t* Bp = (bf16_t*)(ws + apBytes);
  float*  t2 = (float*)(ws + 2 * apBytes);                   // 4 KB
  float*  S  = (float*)(ws + 2 * apBytes + 4096);            // 4 MB
  float*  wn = (float*)(ws + 2 * apBytes + 4096 + (size_t)CP * CP * 4);
  float*  wd = wn + N_SMP;

  msaug_pack_kernel<<<CP, 256, 0, stream>>>(fc, cv, Ap, Bp, t2);

  dim3 g(CP / 64, CP / 64);  // 16 x 16 blocks, 4 waves each
  msaug_gemm_kernel<<<g, 128, 0, stream>>>(Ap, Bp, t2, ratio, S);

  msaug_loss_kernel<<<N_SMP, 256, 0, stream>>>(y_s, labels, S, weights, wn, wd);

  msaug_final_kernel<<<1, 256, 0, stream>>>(wn, wd, (float*)d_out);
}
